// EGNNNet_34479997452471
// MI455X (gfx1250) — matrix operand order repaired and checked by reading, not compiled, not run on hardware
//
#include <hip/hip_runtime.h>

#define C 64
#define Bn 4

typedef __attribute__((ext_vector_type(16))) __bf16 v16bf;
typedef __attribute__((ext_vector_type(8)))  float  v8f;

union BFrag { v16bf v; unsigned short h[16]; };

__device__ __forceinline__ unsigned short f2bf(float f) {
  union { float f; unsigned u; } x; x.f = f;
  unsigned r = x.u + 0x7FFFu + ((x.u >> 16) & 1u);   // round-to-nearest-even
  return (unsigned short)(r >> 16);
}
__device__ __forceinline__ float bf2f(unsigned short h) {
  union { unsigned u; float f; } x; x.u = ((unsigned)h) << 16;
  return x.f;
}
__device__ __forceinline__ float sigmoidf(float x) {
  return 1.0f / (1.0f + __expf(-x));
}
__device__ __forceinline__ v8f wmma_bf16(v16bf a, v16bf b, v8f c) {
  return __builtin_amdgcn_wmma_f32_16x16x32_bf16(false, a, false, b, (short)0, c,
                                                 false, false);
}

// A fragment: 16x32 bf16 tile (rows 0..15 of LDS tile t, K offset kc, row stride ldk).
// ISA layout: lanes 0-15 -> M=lane, K = kc + {0..7, 16..23}; lanes 16-31 -> K += 8.
__device__ __forceinline__ v16bf load_a_frag(const unsigned short* t, int ldk, int kc) {
  int lane = threadIdx.x & 31;
  const unsigned short* p = t + (lane & 15) * ldk + kc + ((lane >> 4) << 3);
  BFrag f;
#pragma unroll
  for (int h = 0; h < 8; ++h) { f.h[h] = p[h]; f.h[8 + h] = p[16 + h]; }
  return f.v;
}
// B fragment: 32x16 bf16 from LDS-transposed weight wt[n*ldk+k] = W[k][n].
// ISA layout: lane holds column n=lane&15; lanes 0-15 K=kc+{0..7,16..23}, lanes 16-31 K += 8.
__device__ __forceinline__ v16bf load_b_frag(const unsigned short* wt, int ldk,
                                             int nt, int kc) {
  int lane = threadIdx.x & 31;
  const unsigned short* p = wt + (nt * 16 + (lane & 15)) * ldk + kc + ((lane >> 4) << 3);
  BFrag f;
#pragma unroll
  for (int h = 0; h < 8; ++h) { f.h[h] = p[h]; f.h[8 + h] = p[16 + h]; }
  return f.v;
}

// Non-returning device-memory float max atomic (STOREcnt; s_endpgm implies wait-idle).
__device__ __forceinline__ void atomic_max_f32(float* p, float v) {
  asm volatile("global_atomic_max_num_f32 %0, %1, off" :: "v"(p), "v"(v) : "memory");
}

// ---------------------------------------------------------------------------
// Generic  out[M,64] = A[M,K] @ W[K,64]  (bf16 WMMA, f32 accumulate).
// CONCAT: logical A = [xp | fix(aggr)] (K=128), both stride-64, -inf -> 0.
// EPI:    out = leaky( resid + acc + bias ), slope 0.01.
// 256 threads = 8 waves; each wave computes one 16x64 tile.
// ---------------------------------------------------------------------------
template<int K, bool CONCAT, bool EPI>
__global__ __launch_bounds__(256) void gemm_n64(
    const float* __restrict__ A, const float* __restrict__ A2,
    const float* __restrict__ W, const float* __restrict__ bias,
    const float* __restrict__ resid, float* __restrict__ out, int M)
{
  constexpr int KP = (K + 31) & ~31;
  __shared__ unsigned short wt[C * KP];
  __shared__ unsigned short at[8][16 * KP];

  const int tid = threadIdx.x, wave = tid >> 5, lane = tid & 31;

  // Transpose + convert weights into LDS (zero-pad K to KP).
  for (int i = tid; i < C * KP; i += 256) {
    int n = i / KP, kk = i % KP;
    float v = (kk < K) ? W[kk * C + n] : 0.0f;
    wt[n * KP + kk] = f2bf(v);
  }
  __syncthreads();

  const int row0 = (blockIdx.x * 8 + wave) * 16;

  // Stage this wave's 16xKP input tile (bf16).
  for (int i = lane; i < 16 * KP; i += 32) {
    int r = i / KP, kk = i % KP;
    int grow = row0 + r;
    float v = 0.0f;
    if (grow < M && kk < K) {
      if (CONCAT) {
        if (kk < C) v = A[grow * C + kk];
        else { float g = A2[grow * C + (kk - C)]; v = (g < -3.0e38f) ? 0.0f : g; }
      } else {
        v = A[grow * K + kk];
      }
    }
    at[wave][r * KP + kk] = f2bf(v);
  }

  v8f acc[4] = {};
#pragma unroll
  for (int kc = 0; kc < KP; kc += 32) {
    v16bf a = load_a_frag(&at[wave][0], KP, kc);
#pragma unroll
    for (int nt = 0; nt < 4; ++nt)
      acc[nt] = wmma_bf16(a, load_b_frag(wt, KP, nt, kc), acc[nt]);
  }

  const int mhi = (lane >> 4) << 3, nlo = lane & 15;
#pragma unroll
  for (int nt = 0; nt < 4; ++nt) {
    int col = nt * 16 + nlo;
#pragma unroll
    for (int v = 0; v < 8; ++v) {
      int row = row0 + mhi + v;
      if (row < M) {
        float r = acc[nt][v];
        if (EPI) {
          r += bias[col] + resid[row * C + col];
          r = (r > 0.0f) ? r : 0.01f * r;
        }
        out[row * C + col] = r;
      }
    }
  }
}

// ---------------------------------------------------------------------------
// Fused edge pipeline: gather xj/xi from L2-resident xp, WMMA qf/kf,
// attention dot + sigmoid gates, scatter segment-max via f32 max atomics.
// 8 waves/block, 4 edges per wave (16 rows = 4 edges x B=4).
// ---------------------------------------------------------------------------
__global__ __launch_bounds__(256) void edge_attention(
    const float* __restrict__ xp, const float* __restrict__ ea,
    const int* __restrict__ eidx,
    const float* __restrict__ q, const float* __restrict__ k,
    const float* __restrict__ aw, const float* __restrict__ ab,
    float* __restrict__ aggr, int E)
{
  __shared__ unsigned short qT[C * C];
  __shared__ unsigned short kT[C * C];
  __shared__ float awv[3 * C];
  __shared__ unsigned short xjb[8][16 * C];
  __shared__ unsigned short xib[8][16 * C];
  __shared__ float eat[8][4 * C];
  __shared__ float qf[8][16 * C];
  __shared__ float kf[8][16 * C];
  __shared__ float attv[8][16];

  const int tid = threadIdx.x, wave = tid >> 5, lane = tid & 31;

  for (int i = tid; i < C * C; i += 256) {
    int n = i >> 6, kk = i & 63;
    qT[n * C + kk] = f2bf(q[kk * C + n]);
    kT[n * C + kk] = f2bf(k[kk * C + n]);
  }
  for (int i = tid; i < 3 * C; i += 256) awv[i] = aw[i];
  __syncthreads();

  const int e0 = (blockIdx.x * 8 + wave) * 4;

  // Stage xj (src gather) and xi (dst gather) as bf16; edge features as f32.
  for (int i = lane; i < 16 * C; i += 32) {
    int r = i >> 6, cc = i & 63;
    int e = e0 + (r >> 2); e = e < E ? e : E - 1;
    int b = r & 3;
    int s = eidx[e];
    int d = eidx[E + e];
    xjb[wave][i] = f2bf(xp[(s * Bn + b) * C + cc]);
    xib[wave][i] = f2bf(xp[(d * Bn + b) * C + cc]);
  }
  for (int i = lane; i < 4 * C; i += 32) {
    int e = e0 + (i >> 6); e = e < E ? e : E - 1;
    eat[wave][i] = ea[e * C + (i & 63)];
  }

  // qf = xj @ q ; kf = xi @ k  (16x64 tiles, K=64 in two 32-chunks)
  v8f aq[4] = {}, ak[4] = {};
#pragma unroll
  for (int kc = 0; kc < C; kc += 32) {
    v16bf a1 = load_a_frag(&xjb[wave][0], C, kc);
    v16bf a2 = load_a_frag(&xib[wave][0], C, kc);
#pragma unroll
    for (int nt = 0; nt < 4; ++nt) {
      aq[nt] = wmma_bf16(a1, load_b_frag(qT, C, nt, kc), aq[nt]);
      ak[nt] = wmma_bf16(a2, load_b_frag(kT, C, nt, kc), ak[nt]);
    }
  }
  const int mhi = (lane >> 4) << 3, nlo = lane & 15;
#pragma unroll
  for (int nt = 0; nt < 4; ++nt)
#pragma unroll
    for (int v = 0; v < 8; ++v) {
      qf[wave][(mhi + v) * C + nt * 16 + nlo] = aq[nt][v];
      kf[wave][(mhi + v) * C + nt * 16 + nlo] = ak[nt][v];
    }

  // att[m] = sigmoid( [qf kf ea] . aw + ab ) ; half-wave split + xor-16 combine.
  {
    int m = lane & 15, half = lane >> 4;
    float s = 0.0f;
#pragma unroll
    for (int c = 0; c < 32; ++c) {
      int cc = half * 32 + c;
      s += qf[wave][m * C + cc] * awv[cc];
      s += kf[wave][m * C + cc] * awv[C + cc];
      s += eat[wave][(m >> 2) * C + cc] * awv[2 * C + cc];
    }
    s += __shfl_xor(s, 16, 32);
    float a = sigmoidf(s + ab[0]);
    if (lane < 16) attv[wave][m] = a;
  }

  // msg = att * xj * sigmoid(ea) ; scatter segment-max into aggr[dst].
  for (int i = lane; i < 16 * C; i += 32) {
    int m = i >> 6, cc = i & 63;
    int e = e0 + (m >> 2);
    if (e < E) {
      float xjv = bf2f(xjb[wave][i]);
      float eav = eat[wave][(m >> 2) * C + cc];
      float msg = attv[wave][m] * xjv * sigmoidf(eav);
      int d = eidx[E + e];
      atomic_max_f32(&aggr[(d * Bn + (m & 3)) * C + cc], msg);
    }
  }
}

__global__ void fill_neginf(float* p, int n) {
  int i = blockIdx.x * 256 + threadIdx.x;
  if (i < n) p[i] = -__builtin_inff();
}

// ---------------------------------------------------------------------------
extern "C" void kernel_launch(void* const* d_in, const int* in_sizes, int n_in,
                              void* d_out, int out_size, void* d_ws, size_t ws_size,
                              hipStream_t stream) {
  const float* X     = (const float*)d_in[0];
  const int*   eidx  = (const int*)  d_in[1];
  const float* eattr = (const float*)d_in[2];
  // layer params, dict order: w_n, w_e, q, k, aw, ab, ow, ob
  const float* w_n1 = (const float*)d_in[3];
  const float* w_e1 = (const float*)d_in[4];
  const float* q1   = (const float*)d_in[5];
  const float* k1   = (const float*)d_in[6];
  const float* aw1  = (const float*)d_in[7];
  const float* ab1  = (const float*)d_in[8];
  const float* ow1  = (const float*)d_in[9];
  const float* ob1  = (const float*)d_in[10];
  const float* w_n2 = (const float*)d_in[11];
  const float* w_e2 = (const float*)d_in[12];
  const float* q2   = (const float*)d_in[13];
  const float* k2   = (const float*)d_in[14];
  const float* aw2  = (const float*)d_in[15];
  const float* ab2  = (const float*)d_in[16];
  const float* ow2  = (const float*)d_in[17];
  const float* ob2  = (const float*)d_in[18];

  const int Nn = in_sizes[0] / (Bn * C);   // 10000 nodes
  const int E  = in_sizes[1] / 2;          // 160000 edges
  const int M  = Nn * Bn;                  // 40000 rows

  char* ws = (char*)d_ws;
  float* xp   = (float*)ws; ws += (size_t)M * C * sizeof(float);
  float* aggr = (float*)ws; ws += (size_t)M * C * sizeof(float);
  float* c1   = (float*)ws; ws += (size_t)M * C * sizeof(float);
  float* ea   = (float*)ws; ws += (size_t)E * C * sizeof(float);
  (void)ws_size; (void)n_in; (void)out_size;

  dim3 blk(256);
  const int gN    = (M + 127) / 128;   // 8 waves x 16 rows per block
  const int gE16  = (E + 127) / 128;
  const int gEdge = (E + 31) / 32;     // 8 waves x 4 edges per block
  const int fillN = M * C;
  const int gFill = (fillN + 255) / 256;

  // ---- layer 1 ----
  gemm_n64<64, false, false><<<gN,   blk, 0, stream>>>(X, nullptr, w_n1, nullptr, nullptr, xp, M);
  gemm_n64<16, false, false><<<gE16, blk, 0, stream>>>(eattr, nullptr, w_e1, nullptr, nullptr, ea, E);
  fill_neginf<<<gFill, blk, 0, stream>>>(aggr, fillN);
  edge_attention<<<gEdge, blk, 0, stream>>>(xp, ea, eidx, q1, k1, aw1, ab1, aggr, E);
  gemm_n64<128, true, true><<<gN, blk, 0, stream>>>(xp, aggr, ow1, ob1, xp, c1, M);

  // ---- layer 2 ----
  gemm_n64<64, false, false><<<gN,   blk, 0, stream>>>(c1, nullptr, w_n2, nullptr, nullptr, xp, M);
  gemm_n64<16, false, false><<<gE16, blk, 0, stream>>>(eattr, nullptr, w_e2, nullptr, nullptr, ea, E);
  fill_neginf<<<gFill, blk, 0, stream>>>(aggr, fillN);
  edge_attention<<<gEdge, blk, 0, stream>>>(xp, ea, eidx, q2, k2, aw2, ab2, aggr, E);
  gemm_n64<128, true, true><<<gN, blk, 0, stream>>>(xp, aggr, ow2, ob2, xp, (float*)d_out, M);
}
